// SqueezeExciteBlock_31860067402274
// MI455X (gfx1250) — matrix operand order repaired and checked by reading, not compiled, not run on hardware
//
#include <hip/hip_runtime.h>

typedef __attribute__((ext_vector_type(16))) _Float16 v16h;
typedef __attribute__((ext_vector_type(8)))  float    v8f;
typedef __attribute__((ext_vector_type(4)))  float    f32x4;   // native vec for NT store

#define B_  32
#define C_  256
#define L_  8192
#define H_  16

// ---------------------------------------------------------------------------
// Pass 1: y[b][c] = mean_L x[b][c][:]
// Streams x front-to-back with regular-temporal b128 loads so the *tail* of x
// (~192 MB) is resident in the global L2 when pass 3 starts.
// ---------------------------------------------------------------------------
__global__ __launch_bounds__(256) void se_mean_kernel(const float* __restrict__ x,
                                                      float* __restrict__ y) {
    const int row = blockIdx.x;                       // b*C + c, 0..8191
    const f32x4* xr = (const f32x4*)(x + (size_t)row * L_);
    const int n4 = L_ / 4;                            // 2048
    float s = 0.f;
    for (int i = threadIdx.x; i < n4; i += 256) {
        f32x4 v = xr[i];
        s += (v.x + v.y) + (v.z + v.w);
    }
    __shared__ float sm[256];
    sm[threadIdx.x] = s;
    __syncthreads();
    #pragma unroll
    for (int off = 128; off > 0; off >>= 1) {
        if (threadIdx.x < off) sm[threadIdx.x] += sm[threadIdx.x + off];
        __syncthreads();
    }
    if (threadIdx.x == 0) y[row] = sm[0] * (1.0f / (float)L_);
}

// ---------------------------------------------------------------------------
// Pass 2: bottleneck MLP on WMMA, one wave32 block.
//   h = relu(y[32,256] @ W1^T[256,16])        16 x v_wmma_f32_16x16x32_f16
//   g = sigmoid(h[32,16] @ W2^T[16,256])      32 x v_wmma_f32_16x16x32_f16
// Fragment layouts per CDNA5 ISA §7.12.2 (wave32).
// ---------------------------------------------------------------------------
union AFrag { v16h v; _Float16 e[16]; };
union CFrag { v8f  v; float    e[8];  };

__global__ __launch_bounds__(32) void se_mlp_wmma_kernel(const float* __restrict__ y,   // [32,256]
                                                         const float* __restrict__ W1,  // [16,256]
                                                         const float* __restrict__ W2,  // [256,16]
                                                         float* __restrict__ g) {       // [32,256]
    const int lane = threadIdx.x;      // 0..31, EXEC all ones
    const int half = lane >> 4;        // 0: lanes 0-15, 1: lanes 16-31
    const int l16  = lane & 15;

    __shared__ _Float16 h16[B_][H_];

    // ---- layer 1: h = relu(y @ W1^T), K = 256 in 8 chunks of 32 ----
    #pragma unroll
    for (int t = 0; t < 2; ++t) {                       // M tiles (rows 0-15, 16-31)
        CFrag acc; acc.v = (v8f){};
        const int row = t * 16 + l16;                   // A: M = lane%16
        for (int kc = 0; kc < 8; ++kc) {
            const int k0 = kc * 32;
            AFrag a, b;
            #pragma unroll
            for (int v = 0; v < 8; ++v) {
                #pragma unroll
                for (int hh = 0; hh < 2; ++hh) {
                    // A (16-bit 16x32): k = (v%4)*2+hh + half*8 + (v/4)*16
                    const int ka = (v & 3) * 2 + hh + half * 8 + (v >> 2) * 16 + k0;
                    a.e[v * 2 + hh] = (_Float16)y[row * C_ + ka];
                    // B (32x16): N = lane%16, k = (v%4)*2+hh + half*8 + (v/4)*16 ; B[k][n] = W1[n][k]
                    const int kb = (v & 3) * 2 + hh + half * 8 + (v >> 2) * 16 + k0;
                    b.e[v * 2 + hh] = (_Float16)W1[l16 * C_ + kb];
                }
            }
            acc.v = __builtin_amdgcn_wmma_f32_16x16x32_f16(
                false, a.v, false, b.v, (short)0, acc.v, false, false);
        }
        // D layout: VGPR r -> M = half*8 + r, N = lane%16
        #pragma unroll
        for (int r = 0; r < 8; ++r) {
            const int m = t * 16 + half * 8 + r;
            const float hv = acc.e[r];
            h16[m][l16] = (_Float16)(hv > 0.f ? hv : 0.f);
        }
    }
    __syncthreads();

    // ---- layer 2: g = sigmoid(h @ W2^T), K = 16 zero-padded to 32 ----
    #pragma unroll
    for (int t = 0; t < 2; ++t) {
        AFrag a;
        const int row = t * 16 + l16;
        #pragma unroll
        for (int v = 0; v < 8; ++v) {
            #pragma unroll
            for (int hh = 0; hh < 2; ++hh) {
                const int ka = (v & 3) * 2 + hh + half * 8 + (v >> 2) * 16;
                a.e[v * 2 + hh] = (ka < H_) ? h16[row][ka] : (_Float16)0.0f;
            }
        }
        for (int nt = 0; nt < 16; ++nt) {               // N tiles over C=256
            AFrag b;
            const int n = nt * 16 + l16;                // global output channel
            #pragma unroll
            for (int v = 0; v < 8; ++v) {
                #pragma unroll
                for (int hh = 0; hh < 2; ++hh) {
                    const int kb = (v & 3) * 2 + hh + half * 8 + (v >> 2) * 16;
                    b.e[v * 2 + hh] = (kb < H_) ? (_Float16)W2[n * H_ + kb]
                                                : (_Float16)0.0f;
                }
            }
            CFrag acc; acc.v = (v8f){};
            acc.v = __builtin_amdgcn_wmma_f32_16x16x32_f16(
                false, a.v, false, b.v, (short)0, acc.v, false, false);
            #pragma unroll
            for (int r = 0; r < 8; ++r) {
                const int m = t * 16 + half * 8 + r;
                const float gv = 1.0f / (1.0f + __expf(-acc.e[r]));
                g[m * C_ + nt * 16 + l16] = gv;
            }
        }
    }
}

// ---------------------------------------------------------------------------
// Pass 3: out = x * g[b][c]
// Blocks sweep x TAIL-to-HEAD in contiguous 16 KB chunks so the pass starts on
// the ~192 MB of x still resident in the 192 MB global L2 from pass 1 (front-
// to-back pass 1 + LRU leaves the tail cached).  Output uses non-temporal
// stores (TH=NT) so the 256 MB writeback does not evict unread x lines.
// ---------------------------------------------------------------------------
#define SCALE_BLOCKS 16384
#define CHUNK4 1024                     // float4s per block: 16 KB, 4 iters/thread

__global__ __launch_bounds__(256) void se_scale_kernel(const f32x4* __restrict__ x4,
                                                       const float* __restrict__ g,
                                                       f32x4* __restrict__ out4) {
    // reverse chunk mapping: block 0 takes the last chunk
    const unsigned chunk = (SCALE_BLOCKS - 1u) - blockIdx.x;
    const unsigned base  = chunk * CHUNK4;
    #pragma unroll
    for (int it = 0; it < CHUNK4 / 256; ++it) {
        const unsigned i = base + it * 256 + threadIdx.x;
        const unsigned row = i >> 11;               // i / (L/4=2048) -> b*C + c
        const float gv = g[row];
        f32x4 v = x4[i];
        v.x *= gv; v.y *= gv; v.z *= gv; v.w *= gv;
        __builtin_nontemporal_store(v, &out4[i]);   // NT store: don't pollute L2
    }
}

// ---------------------------------------------------------------------------
extern "C" void kernel_launch(void* const* d_in, const int* in_sizes, int n_in,
                              void* d_out, int out_size, void* d_ws, size_t ws_size,
                              hipStream_t stream) {
    const float* x  = (const float*)d_in[0];   // [32,256,8192]
    const float* W1 = (const float*)d_in[1];   // [16,256]
    const float* W2 = (const float*)d_in[2];   // [256,16]
    float* out = (float*)d_out;

    float* y = (float*)d_ws;                   // [32,256] = 32 KB
    float* g = y + (size_t)B_ * C_;            // [32,256] = 32 KB

    se_mean_kernel<<<B_ * C_, 256, 0, stream>>>(x, y);
    se_mlp_wmma_kernel<<<1, 32, 0, stream>>>(y, W1, W2, g);
    se_scale_kernel<<<SCALE_BLOCKS, 256, 0, stream>>>((const f32x4*)x, g, (f32x4*)out);
}
